// Correlation_Layer_48687749268009
// MI455X (gfx1250) — matrix operand order repaired and checked by reading, not compiled, not run on hardware
//
#include <hip/hip_runtime.h>

typedef unsigned int   u32;
typedef unsigned short u16;

typedef __attribute__((ext_vector_type(16))) __bf16 v16bf;
typedef __attribute__((ext_vector_type(8)))  float  v8f;

#define NIMG 6
#define NCH  512
#define HIN  90
#define H2   45
#define HW   2025      // 45*45
#define HWP  2032      // padded hw stride (multiple of 8 -> 16B-aligned b128 loads)
#define NPAIR 12
#define BM   128
#define BN   128
#define BK   32
#define NK   (NCH / BK)   // 16 k-steps
#define LDK  40           // padded K stride in LDS (elements); row = 80B, multiple of 16B

// ---------------- Stage 1: 2x2 average pool + f32 -> bf16 (RNE) ----------------
__global__ __launch_bounds__(256) void pool_bf16_kernel(const float* __restrict__ x,
                                                        u16* __restrict__ xf) {
  int idx = blockIdx.x * 256 + threadIdx.x;
  if (idx >= NIMG * NCH * HW) return;
  int q   = idx % HW;
  int ch  = (idx / HW) % NCH;
  int img = idx / (HW * NCH);
  int h2 = q / H2, w2 = q % H2;
  const float* p = x + ((size_t)(img * NCH + ch) * HIN + 2 * h2) * HIN + 2 * w2;
  // x is read-once: non-temporal loads keep L2 free for the pooled tensor.
  float s = (__builtin_nontemporal_load(p)       + __builtin_nontemporal_load(p + 1) +
             __builtin_nontemporal_load(p + HIN) + __builtin_nontemporal_load(p + HIN + 1)) * 0.25f;
  u32 u = __float_as_uint(s);
  u += 0x7FFFu + ((u >> 16) & 1u);           // round-to-nearest-even to bf16
  xf[(size_t)(img * NCH + ch) * HWP + q] = (u16)(u >> 16);   // regular: want L2-resident
}

// ---------------- Stage 2: batched C = A^T * B via bf16 WMMA ----------------
// xf layout: [img(6)][c(512)][HWP], bf16 bits. Block computes a BMxBN tile of
// corr[pair]; 8 waves, each wave a 64x32 subtile (4x2 grid of 16x16 WMMA tiles).
// LDS is double-buffered: one barrier per k-step.
__global__ __launch_bounds__(256) void corr_wmma_kernel(const u16* __restrict__ xf,
                                                        float* __restrict__ out) {
  __shared__ u16 lds[2][(BM + BN) * LDK];   // [buf][ A: [m][k] | B: [n][k] ]

  const int t     = threadIdx.x;
  const int lane  = t & 31;
  const int wave  = t >> 5;        // 0..7
  const int waveM = wave & 1;      // 2 row-groups of 64
  const int waveN = wave >> 1;     // 4 col-groups of 32
  const int l16   = lane & 15;
  const int half  = lane >> 4;

  const int qtile = blockIdx.x * BM;
  const int ptile = blockIdx.y * BN;
  const int pair  = blockIdx.z;    // 0..11

  // pair -> (batch, imgA, imgB); idx order: (0,1)(0,2)(1,0)(1,2)(2,0)(2,1)
  const int b   = pair / 6;
  const int k6  = pair % 6;
  const int i_  = k6 >> 1;
  const int js  = k6 & 1;
  const int j_  = (i_ == 0) ? (js ? 2 : 1) : (i_ == 1) ? (js ? 2 : 0) : (js ? 1 : 0);

  const u16* __restrict__ Abase = xf + (size_t)(b * 3 + i_) * NCH * HWP;
  const u16* __restrict__ Bbase = xf + (size_t)(b * 3 + j_) * NCH * HWP;

  // Staging roles: threads 0..127 stage the A chunk, 128..255 the B chunk.
  // Per chunk & k-step: 16 hw-segments (8 elems) x 8 k-groups (4 ch) = 128 threads.
  const int  tt       = t & 127;
  const int  seg      = tt >> 3;              // 0..15
  const int  g        = tt & 7;               // 0..7
  const bool isB      = (t >= 128);
  const u16* src      = isB ? Bbase : Abase;
  const int  chunkOff = isB ? BM * LDK : 0;
  const int  qbase    = (isB ? ptile : qtile) + seg * 8;

  // global -> registers: 8 q x 4 consecutive c per thread
  auto loadChunk = [&](int k0, uint4 (&vals)[4]) {
    if (qbase + 8 <= HW) {
#pragma unroll
      for (int cc = 0; cc < 4; ++cc)
        vals[cc] = *reinterpret_cast<const uint4*>(src + (size_t)(k0 + g * 4 + cc) * HWP + qbase);
    } else {
#pragma unroll
      for (int cc = 0; cc < 4; ++cc) {
        const u16* r = src + (size_t)(k0 + g * 4 + cc) * HWP;
        u32 w[4] = {0u, 0u, 0u, 0u};
#pragma unroll
        for (int jj = 0; jj < 8; ++jj) {
          u32 e = (qbase + jj < HW) ? (u32)r[qbase + jj] : 0u;
          w[jj >> 1] |= e << (16 * (jj & 1));
        }
        vals[cc] = make_uint4(w[0], w[1], w[2], w[3]);
      }
    }
  };

  // registers -> LDS transposed [hw][k]: pack 4 consecutive k per b64 store
  auto storeChunk = [&](u16* dstbuf, const uint4 (&vals)[4]) {
#pragma unroll
    for (int jj = 0; jj < 8; ++jj) {
      const u32 sh = 16u * (jj & 1);
      u32 e0 = (((const u32*)&vals[0])[jj >> 1] >> sh) & 0xFFFFu;
      u32 e1 = (((const u32*)&vals[1])[jj >> 1] >> sh) & 0xFFFFu;
      u32 e2 = (((const u32*)&vals[2])[jj >> 1] >> sh) & 0xFFFFu;
      u32 e3 = (((const u32*)&vals[3])[jj >> 1] >> sh) & 0xFFFFu;
      int row = seg * 8 + jj;
      *reinterpret_cast<uint2*>(dstbuf + row * LDK + g * 4) =
          make_uint2(e0 | (e1 << 16), e2 | (e3 << 16));
    }
  };

  // ---- prologue: stage chunk 0 into buffer 0 ----
  uint4 vals[4];
  loadChunk(0, vals);
  storeChunk(&lds[0][chunkOff], vals);
  __syncthreads();

  v8f acc[4][2] = {};

  for (int kk = 0; kk < NK; ++kk) {
    const u16* ldsA = &lds[kk & 1][0];
    const u16* ldsB = &lds[kk & 1][BM * LDK];
    const bool more = (kk + 1 < NK);

    // prefetch next chunk (global latency hides under this step's WMMAs)
    if (more) loadChunk((kk + 1) * BK, vals);

    // fragment loads (ISA 16-bit A 16x32 / B 32x16 lane layouts)
    union Frag { uint4 u[2]; v16bf v; };
    Frag a[4], bb[2];
#pragma unroll
    for (int i = 0; i < 4; ++i) {
      // A: lane<16 holds M=l16, K = {hs*8+0..7} U {16+hs*8+0..7}
      const u16* pA = ldsA + (waveM * 64 + i * 16 + l16) * LDK + half * 8;
      a[i].u[0] = *reinterpret_cast<const uint4*>(pA);
      a[i].u[1] = *reinterpret_cast<const uint4*>(pA + 16);
    }
#pragma unroll
    for (int j = 0; j < 2; ++j) {
      // B: lane holds N=l16, K = {half*8+0..7} U {16+half*8+0..7}
      const u16* pB = ldsB + (waveN * 32 + j * 16 + l16) * LDK + half * 8;
      bb[j].u[0] = *reinterpret_cast<const uint4*>(pB);
      bb[j].u[1] = *reinterpret_cast<const uint4*>(pB + 16);
    }

    // scatter next chunk into the other buffer (no conflict with current reads)
    if (more) storeChunk(&lds[(kk + 1) & 1][chunkOff], vals);

#pragma unroll
    for (int i = 0; i < 4; ++i)
#pragma unroll
      for (int j = 0; j < 2; ++j)
        acc[i][j] = __builtin_amdgcn_wmma_f32_16x16x32_bf16(
            false, a[i].v, false, bb[j].v, (short)0, acc[i][j], false, false);

    __syncthreads();   // one barrier per k-step (ping-pong)
  }

  // ---- epilogue: C/D layout lane=N col, VGPR v -> M = v + 8*half ----
  // Output is written once and never re-read: non-temporal stores keep the
  // 197MB stream from evicting the L2-resident 12.5MB pooled tensor.
  float* __restrict__ obase = out + (size_t)pair * HW * HW;
#pragma unroll
  for (int i = 0; i < 4; ++i) {
    int qb = qtile + waveM * 64 + i * 16 + half * 8;
#pragma unroll
    for (int j = 0; j < 2; ++j) {
      int p = ptile + waveN * 32 + j * 16 + l16;
      if (p < HW) {
#pragma unroll
        for (int v = 0; v < 8; ++v) {
          int q = qb + v;
          if (q < HW)
            __builtin_nontemporal_store(acc[i][j][v], obase + (size_t)q * HW + p);
        }
      }
    }
  }
}

extern "C" void kernel_launch(void* const* d_in, const int* in_sizes, int n_in,
                              void* d_out, int out_size, void* d_ws, size_t ws_size,
                              hipStream_t stream) {
  const float* x = (const float*)d_in[0];   // (6,512,90,90) f32; d_in[1] (n=3) is fixed
  float* out = (float*)d_out;               // (12, 2025, 45, 45) f32 == 12 x 2025 x 2025
  u16* xf = (u16*)d_ws;                     // bf16 pooled tensor: 6*512*2032*2 B ~ 12.5 MB

  if (ws_size < (size_t)NIMG * NCH * HWP * sizeof(u16)) return;

  const int total = NIMG * NCH * HW;
  pool_bf16_kernel<<<(total + 255) / 256, 256, 0, stream>>>(x, xf);

  dim3 grid((HW + BM - 1) / BM, (HW + BN - 1) / BN, NPAIR);
  corr_wmma_kernel<<<grid, 256, 0, stream>>>(xf, out);
}
